// DANetwork_2894807957848
// MI455X (gfx1250) — matrix operand order repaired and checked by reading, not compiled, not run on hardware
//
#include <hip/hip_runtime.h>
#include <hip/hip_bf16.h>

// ---------------------------------------------------------------------------
// DARTS-like 3-cell network forward for MI455X (gfx1250).
// Working set ~45MB (fits in 192MB L2) => latency bound; strategy:
//  - 1x1 convs (the only GEMMs: M=16384, K<=128, N<=64) via v_wmma_f32_16x16x32_f16,
//    operands pre-packed into the exact CDNA5 fragment layout so the GEMM inner
//    loop is pure b128 loads + wmma (no predication, no per-element waits)
//  - mixed spatial ops fused per 32x32 plane in LDS, plane filled with
//    global_load_async_to_lds_b128 (ASYNCcnt path)
//  - cell "concat" is free: step k writes channels [k*C,(k+1)*C) of the cell buffer
// ---------------------------------------------------------------------------

typedef __attribute__((ext_vector_type(16))) _Float16 v16h;
typedef __attribute__((ext_vector_type(8)))  float    v8f;

#define NB 16      // batch
#define HW 1024    // 32*32

__constant__ float kCos[6] = { 0.70710678f, 0.0f, -0.70710678f, -1.0f, -0.70710678f,  0.0f };
__constant__ float kSin[6] = { 0.70710678f, 1.0f,  0.70710678f,  0.0f, -0.70710678f, -1.0f };

// ---- collapse softmax(alphas) -> 11 primitive-transform coefficients -------
__global__ void coeff_kernel(const float* __restrict__ an, const float* __restrict__ ar,
                             float* __restrict__ coeffs) {
  int t = threadIdx.x;
  if (t >= 28) return;
  int set = t / 14, row = t % 14;
  const float* a = (set == 0 ? an : ar) + row * 27;
  float mx = a[0];
  for (int i = 1; i < 27; ++i) mx = fmaxf(mx, a[i]);
  float w[27], s = 0.f;
  for (int i = 0; i < 27; ++i) { w[i] = __expf(a[i] - mx); s += w[i]; }
  float inv = 1.f / s;
  for (int i = 0; i < 27; ++i) w[i] *= inv;
  const float pb[5]  = {0.f, .3f, .4f, .6f, .8f};   // blur probs (note p0.2 builds 0.3)
  const float pi_[4] = {0.f, .2f, .6f, .8f};        // invert probs
  const float pf[5]  = {0.f, .2f, .4f, .6f, .8f};   // flip probs
  float c[11];
  for (int i = 0; i < 11; ++i) c[i] = 0.f;
  c[0] = w[0] + w[20];                               // 'none' + rotate_0
  for (int i = 0; i < 5; ++i) { c[1] += pb[i]*w[1+i];   c[0] += (1.f-pb[i])*w[1+i]; }
  for (int i = 0; i < 4; ++i) { c[2] += pi_[i]*w[6+i];  c[0] += (1.f-pi_[i])*w[6+i]; }
  for (int i = 0; i < 5; ++i) { c[3] += pf[i]*w[10+i];  c[0] += (1.f-pf[i])*w[10+i]; }
  for (int i = 0; i < 5; ++i) { c[4] += pf[i]*w[15+i];  c[0] += (1.f-pf[i])*w[15+i]; }
  for (int j = 0; j < 6; ++j) c[5+j] = w[21+j];
  float* o = coeffs + (size_t)(set*14 + row) * 11;
  for (int i = 0; i < 11; ++i) o[i] = c[i];
}

// ---- stem: 3x3 conv 3->48, zero pad ----------------------------------------
__global__ void stem_conv_kernel(const float* __restrict__ x, const float* __restrict__ w,
                                 float* __restrict__ out) {
  int idx = blockIdx.x * blockDim.x + threadIdx.x;   // over NB*48*HW
  if (idx >= NB * 48 * HW) return;
  int hw = idx & (HW - 1);
  int co = (idx >> 10) % 48;
  int n  = idx / (48 * HW);
  int y = hw >> 5, xw = hw & 31;
  float acc = 0.f;
  for (int ci = 0; ci < 3; ++ci) {
    const float* xp = x + (size_t)(n * 3 + ci) * HW;
    const float* wp = w + (size_t)(co * 3 + ci) * 9;
    #pragma unroll
    for (int ky = 0; ky < 3; ++ky) {
      int yy = y + ky - 1;
      if ((unsigned)yy >= 32u) continue;
      #pragma unroll
      for (int kx = 0; kx < 3; ++kx) {
        int xx = xw + kx - 1;
        if ((unsigned)xx >= 32u) continue;
        acc = fmaf(xp[yy * 32 + xx], wp[ky * 3 + kx], acc);
      }
    }
  }
  out[idx] = acc;
}

// ---- training-mode batchnorm: stats over (N,H,W) per channel ---------------
__global__ void bn_stats_kernel(const float* __restrict__ buf, int C,
                                float* __restrict__ mean, float* __restrict__ rstd) {
  int c = blockIdx.x;
  float s = 0.f, ss = 0.f;
  for (int i = threadIdx.x; i < NB * HW; i += blockDim.x) {
    int n = i >> 10, hw = i & 1023;
    float v = buf[((size_t)n * C + c) * HW + hw];
    s += v; ss += v * v;
  }
  __shared__ float sh[256], sh2[256];
  sh[threadIdx.x] = s; sh2[threadIdx.x] = ss;
  __syncthreads();
  for (int off = 128; off > 0; off >>= 1) {
    if (threadIdx.x < off) {
      sh[threadIdx.x]  += sh[threadIdx.x + off];
      sh2[threadIdx.x] += sh2[threadIdx.x + off];
    }
    __syncthreads();
  }
  if (threadIdx.x == 0) {
    float m = sh[0] * (1.f / (NB * HW));
    float v = sh2[0] * (1.f / (NB * HW)) - m * m;
    mean[c] = m;
    rstd[c] = rsqrtf(v + 1e-5f);
  }
}

__global__ void bn_apply_kernel(float* __restrict__ buf, int C,
                                const float* __restrict__ mean, const float* __restrict__ rstd,
                                const float* __restrict__ gamma, const float* __restrict__ beta) {
  int idx = blockIdx.x * blockDim.x + threadIdx.x;
  if (idx >= NB * C * HW) return;
  int c = (idx >> 10) % C;
  float v = (buf[idx] - mean[c]) * rstd[c];
  if (gamma) v = v * gamma[c] + beta[c];
  buf[idx] = v;
}

// ---- pack ReLU(activations) into WMMA A-fragment layout ---------------------
// apk[((tm*KC + kc)*32 + lane)*16 + t]  (ISA 7.12.2, 16-bit A 16x32)
__global__ void pack_a_kernel(const float* __restrict__ in, int Cin, int KC,
                              _Float16* __restrict__ apk) {
  int idx = blockIdx.x * blockDim.x + threadIdx.x;   // over 1024*KC*512
  if (idx >= 1024 * KC * 512) return;
  int t    = idx & 15;
  int lane = (idx >> 4) & 31;
  int kc   = (idx >> 9) % KC;
  int tm   = (idx >> 9) / KC;
  int kA = (lane < 16) ? 0 : 8;
  int k  = (kc << 5) + kA + t + ((t >= 8) ? 8 : 0);
  int m  = tm * 16 + (lane & 15);
  float v = 0.f;
  if (k < Cin) v = fmaxf(in[((size_t)(m >> 10) * Cin + k) * HW + (m & 1023)], 0.f);
  apk[idx] = (_Float16)v;
}

// ---- pack weights into WMMA B-fragment layout -------------------------------
// bpk[((tn*KC + kc)*32 + lane)*16 + t]  (ISA 7.12.2, 16-bit B 32x16)
__global__ void pack_b_kernel(const float* __restrict__ w, int Cin, int KC, int Cout,
                              _Float16* __restrict__ bpk) {
  int idx = blockIdx.x * blockDim.x + threadIdx.x;   // over tilesN*KC*512
  int tilesN = Cout >> 4;
  if (idx >= tilesN * KC * 512) return;
  int t    = idx & 15;
  int lane = (idx >> 4) & 31;
  int kc   = (idx >> 9) % KC;
  int tn   = (idx >> 9) / KC;
  int kB = (lane < 16) ? 0 : 8;
  int k  = (kc << 5) + kB + t + ((t >= 8) ? 8 : 0);
  int co = tn * 16 + (lane & 15);
  float v = (k < Cin) ? w[(size_t)co * Cin + k] : 0.f;
  bpk[idx] = (_Float16)v;
}

// ---- 1x1 conv GEMM: pure fragment loads + WMMA ------------------------------
// One wave owns 4 consecutive M-tiles for a fixed N-tile (B reused 4x).
__global__ void pre_gemm_kernel(const _Float16* __restrict__ apk,
                                const _Float16* __restrict__ bpk,
                                int KC, float* __restrict__ out, int Cout) {
  const int lane = threadIdx.x & 31;
  const int wid  = blockIdx.x * (blockDim.x >> 5) + (threadIdx.x >> 5);
  const int tilesN = Cout >> 4;
  const int tn = wid % tilesN;
  const int tg = wid / tilesN;                 // 0..255 -> m-tiles 4*tg..4*tg+3
  v8f acc[4] = {};
  const int astride = KC * 512;                // halfs per m-tile
  for (int kc = 0; kc < KC; ++kc) {
    v16h b = *(const v16h*)(bpk + (((size_t)tn * KC + kc) * 32 + lane) * 16);
    const _Float16* abase = apk + (((size_t)(tg * 4) * KC + kc) * 32 + lane) * 16;
    v16h a0 = *(const v16h*)(abase);
    v16h a1 = *(const v16h*)(abase + astride);
    v16h a2 = *(const v16h*)(abase + 2 * astride);
    v16h a3 = *(const v16h*)(abase + 3 * astride);
    acc[0] = __builtin_amdgcn_wmma_f32_16x16x32_f16(false, a0, false, b, (short)0, acc[0], false, false);
    acc[1] = __builtin_amdgcn_wmma_f32_16x16x32_f16(false, a1, false, b, (short)0, acc[1], false, false);
    acc[2] = __builtin_amdgcn_wmma_f32_16x16x32_f16(false, a2, false, b, (short)0, acc[2], false, false);
    acc[3] = __builtin_amdgcn_wmma_f32_16x16x32_f16(false, a3, false, b, (short)0, acc[3], false, false);
  }
  const int co = tn * 16 + (lane & 15);
  #pragma unroll
  for (int i = 0; i < 4; ++i) {
    int mbase = (tg * 4 + i) * 16 + ((lane < 16) ? 0 : 8);
    #pragma unroll
    for (int r = 0; r < 8; ++r) {
      int m = mbase + r;
      out[((size_t)(m >> 10) * Cout + co) * HW + (m & 1023)] = acc[i][r];
    }
  }
}

// ---- one DAG step: new_state = sum_j MixedOp(state_j, coeff_j) --------------
struct StepArgs {
  const float* in[5];
  int nstride[5];        // per-image element stride of each input buffer
  int n_inputs;
  const float* coeff;    // n_inputs * 11 coefficients
  float* out;            // base of this state's channel slab in the cell buffer
  int out_nstride;       // 4*C*HW (concat layout)
  int C;                 // channels per state
};

__global__ void mixed_step_kernel(StepArgs args) {
  __shared__ float pl[HW];
  const int n = blockIdx.x / args.C;
  const int c = blockIdx.x % args.C;
  const int t = threadIdx.x;
  const unsigned ldsdst = (unsigned)(uintptr_t)(&pl[0]) + (unsigned)t * 16u;
  float acc[4] = {0.f, 0.f, 0.f, 0.f};
  for (int j = 0; j < args.n_inputs; ++j) {
    const float* src = args.in[j] + (size_t)n * args.nstride[j] + (size_t)c * HW;
    __syncthreads();   // previous iteration's readers are done with LDS
    {
      // async fill: 256 lanes x 16B = 4KB plane  (GLOBAL_LOAD_ASYNC_TO_LDS_B128)
      unsigned long long ga = (unsigned long long)(const void*)(src + (size_t)t * 4);
      asm volatile("global_load_async_to_lds_b128 %0, %1, off"
                   :: "v"(ldsdst), "v"(ga) : "memory");
      asm volatile("s_wait_asynccnt 0x0" ::: "memory");
    }
    __syncthreads();
    const float* cf = args.coeff + j * 11;
    float c0 = cf[0], c1 = cf[1], c2 = cf[2], c3 = cf[3], c4 = cf[4];
    float cr[6];
    #pragma unroll
    for (int r = 0; r < 6; ++r) cr[r] = cf[5 + r];
    #pragma unroll
    for (int q = 0; q < 4; ++q) {
      int p = t + 256 * q;
      int y = p >> 5, x = p & 31;
      float v = pl[p];
      float r = (c0 - c2) * v + c2;                       // identity + invert
      float bs = 0.f;                                     // 3x3 box blur, reflect pad
      #pragma unroll
      for (int dy = -1; dy <= 1; ++dy) {
        int yy = y + dy; yy = (yy < 0) ? -yy : ((yy > 31) ? 62 - yy : yy);
        #pragma unroll
        for (int dx = -1; dx <= 1; ++dx) {
          int xx = x + dx; xx = (xx < 0) ? -xx : ((xx > 31) ? 62 - xx : xx);
          bs += pl[yy * 32 + xx];
        }
      }
      r = fmaf(c1, bs * (1.f / 9.f), r);
      r = fmaf(c3, pl[y * 32 + (31 - x)], r);             // hflip
      r = fmaf(c4, pl[(31 - y) * 32 + x], r);             // vflip
      #pragma unroll
      for (int a = 0; a < 6; ++a) {                       // bilinear rotations
        float ct = kCos[a], st = kSin[a];
        float fx = (float)x - 15.5f, fy = (float)y - 15.5f;
        float xs = ct * fx - st * fy + 15.5f;
        float ys = st * fx + ct * fy + 15.5f;
        float x0f = floorf(xs), y0f = floorf(ys);
        int x0 = (int)x0f, y0 = (int)y0f;
        float wx1 = xs - x0f, wx0 = 1.f - wx1;
        float wy1 = ys - y0f, wy0 = 1.f - wy1;
        float g = 0.f;
        if ((unsigned)y0 < 32u) {
          if ((unsigned)x0       < 32u) g = fmaf(pl[y0 * 32 + x0],       wy0 * wx0, g);
          if ((unsigned)(x0 + 1) < 32u) g = fmaf(pl[y0 * 32 + x0 + 1],   wy0 * wx1, g);
        }
        if ((unsigned)(y0 + 1) < 32u) {
          if ((unsigned)x0       < 32u) g = fmaf(pl[(y0+1) * 32 + x0],     wy1 * wx0, g);
          if ((unsigned)(x0 + 1) < 32u) g = fmaf(pl[(y0+1) * 32 + x0 + 1], wy1 * wx1, g);
        }
        r = fmaf(cr[a], g, r);
      }
      acc[q] += r;
    }
  }
  float* op = args.out + (size_t)n * args.out_nstride + (size_t)c * HW;
  #pragma unroll
  for (int q = 0; q < 4; ++q) op[t + 256 * q] = acc[q];
}

// ---- head: 3x3 conv 256->3 + bias, zero pad --------------------------------
__global__ void out_conv_kernel(const float* __restrict__ in, const float* __restrict__ w,
                                const float* __restrict__ b, float* __restrict__ out) {
  int idx = blockIdx.x * blockDim.x + threadIdx.x;  // over NB*HW
  if (idx >= NB * HW) return;
  int n = idx >> 10, hw = idx & 1023;
  int y = hw >> 5, x = hw & 31;
  float a0 = b[0], a1 = b[1], a2 = b[2];
  const float* ip = in + (size_t)n * 256 * HW;
  for (int ci = 0; ci < 256; ++ci) {
    const float* pc = ip + (size_t)ci * HW;
    if (ci + 1 < 256) __builtin_prefetch(pc + HW, 0, 1);   // global_prefetch_b8
    #pragma unroll
    for (int ky = 0; ky < 3; ++ky) {
      int yy = y + ky - 1;
      if ((unsigned)yy >= 32u) continue;
      #pragma unroll
      for (int kx = 0; kx < 3; ++kx) {
        int xx = x + kx - 1;
        if ((unsigned)xx >= 32u) continue;
        float v = pc[yy * 32 + xx];
        int widx = ci * 9 + ky * 3 + kx;                   // OIHW, co stride 2304
        a0 = fmaf(v, w[widx],        a0);
        a1 = fmaf(v, w[2304 + widx], a1);
        a2 = fmaf(v, w[4608 + widx], a2);
      }
    }
  }
  float* op = out + (size_t)n * 3 * HW + hw;
  op[0] = a0; op[HW] = a1; op[2 * HW] = a2;
}

// ---------------------------------------------------------------------------
extern "C" void kernel_launch(void* const* d_in, const int* in_sizes, int n_in,
                              void* d_out, int out_size, void* d_ws, size_t ws_size,
                              hipStream_t stream) {
  const float* x             = (const float*)d_in[0];
  const float* stem_w        = (const float*)d_in[1];
  const float* stem_gamma    = (const float*)d_in[2];
  const float* stem_beta     = (const float*)d_in[3];
  const float* pre0_w0       = (const float*)d_in[4];
  const float* pre1_w0       = (const float*)d_in[5];
  const float* pre0_w1       = (const float*)d_in[6];
  const float* pre1_w1       = (const float*)d_in[7];
  const float* pre0_w2       = (const float*)d_in[8];
  const float* pre1_w2       = (const float*)d_in[9];
  const float* alphas_normal = (const float*)d_in[10];
  const float* alphas_reduce = (const float*)d_in[11];
  const float* out_w         = (const float*)d_in[12];
  const float* out_b         = (const float*)d_in[13];
  float* outp = (float*)d_out;

  // workspace layout (floats); ~11.3M floats (~45 MB)
  float* ws = (float*)d_ws;
  size_t o = 0;
  float* buf_stem = ws + o; o += (size_t)NB * 48  * HW;
  float* s0p      = ws + o; o += (size_t)NB * 64  * HW;
  float* s1p      = ws + o; o += (size_t)NB * 64  * HW;
  float* cell0    = ws + o; o += (size_t)NB * 64  * HW;
  float* cell1    = ws + o; o += (size_t)NB * 128 * HW;
  float* cell2    = ws + o; o += (size_t)NB * 256 * HW;
  _Float16* apk   = (_Float16*)(ws + o); o += (size_t)1024 * 4 * 512 / 2;  // 2M halfs
  _Float16* bpk   = (_Float16*)(ws + o); o += (size_t)4 * 4 * 512 / 2;     // 8K halfs
  float* bn_mean  = ws + o; o += 256;
  float* bn_rstd  = ws + o; o += 256;
  float* coeffs   = ws + o; o += 2 * 14 * 11;
  (void)ws_size; (void)in_sizes; (void)n_in; (void)out_size;

  coeff_kernel<<<1, 32, 0, stream>>>(alphas_normal, alphas_reduce, coeffs);

  stem_conv_kernel<<<(NB * 48 * HW + 255) / 256, 256, 0, stream>>>(x, stem_w, buf_stem);
  bn_stats_kernel<<<48, 256, 0, stream>>>(buf_stem, 48, bn_mean, bn_rstd);
  bn_apply_kernel<<<(NB * 48 * HW + 255) / 256, 256, 0, stream>>>(
      buf_stem, 48, bn_mean, bn_rstd, stem_gamma, stem_beta);

  auto run_pre = [&](const float* src, int Cin, const float* w, float* dst, int Cout) {
    const int KC = (Cin + 31) >> 5;
    const int tilesN = Cout >> 4;
    pack_a_kernel<<<(1024 * KC * 512 + 255) / 256, 256, 0, stream>>>(src, Cin, KC, apk);
    pack_b_kernel<<<(tilesN * KC * 512 + 255) / 256, 256, 0, stream>>>(w, Cin, KC, Cout, bpk);
    pre_gemm_kernel<<<64 * tilesN, 128, 0, stream>>>(apk, bpk, KC, dst, Cout);
    bn_stats_kernel<<<Cout, 256, 0, stream>>>(dst, Cout, bn_mean, bn_rstd);
    bn_apply_kernel<<<(NB * Cout * HW + 255) / 256, 256, 0, stream>>>(
        dst, Cout, bn_mean, bn_rstd, nullptr, nullptr);
  };

  auto run_cell = [&](const float* s0src, int C0, const float* s1src, int C1,
                      const float* p0w, const float* p1w, int C, float* cellout, int set) {
    run_pre(s0src, C0, p0w, s0p, C);
    run_pre(s1src, C1, p1w, s1p, C);
    const int offs[4] = {0, 2, 5, 9};
    for (int step = 0; step < 4; ++step) {
      StepArgs a{};
      a.in[0] = s0p; a.nstride[0] = C * HW;
      a.in[1] = s1p; a.nstride[1] = C * HW;
      for (int k = 0; k < step; ++k) {
        a.in[2 + k]      = cellout + (size_t)k * C * HW;   // prior states live in
        a.nstride[2 + k] = 4 * C * HW;                     // the concat buffer
      }
      a.n_inputs    = 2 + step;
      a.coeff       = coeffs + (size_t)(set * 14 + offs[step]) * 11;
      a.out         = cellout + (size_t)step * C * HW;
      a.out_nstride = 4 * C * HW;
      a.C           = C;
      mixed_step_kernel<<<NB * C, 256, 0, stream>>>(a);
    }
  };

  // s0,s1 = stem,stem -> cell0(C=16,out 64ch, normal alphas)
  run_cell(buf_stem, 48, buf_stem, 48, pre0_w0, pre1_w0, 16, cell0, 0);
  // s0,s1 = stem,cell0 -> cell1(C=32,out 128ch, reduce alphas)
  run_cell(buf_stem, 48, cell0, 64, pre0_w1, pre1_w1, 32, cell1, 1);
  // s0,s1 = cell0,cell1 -> cell2(C=64,out 256ch, reduce alphas)
  run_cell(cell0, 64, cell1, 128, pre0_w2, pre1_w2, 64, cell2, 1);

  out_conv_kernel<<<NB * HW / 256, 256, 0, stream>>>(cell2, out_w, out_b, outp);
}
